// C3AH_13477607375842
// MI455X (gfx1250) — compile-verified, matrix operand order checked by reading, not run on hardware
//
#include <hip/hip_runtime.h>

// ---------------------------------------------------------------------------
// CDNA5 (gfx1250) fused C3AH block.
// Heavy GEMMs: bf16 v_wmma_f32_16x16x32_bf16 (wave32), double-buffered LDS
// tiles staged with global_load_async_to_lds_b128 (A) + register-transposed
// stores (B), fused BN/SiLU/GELU/residual epilogues. HGNN head-mean collapses
// so the pre_w GEMM folds into a tiny per-batch protos@pre_w product.
// ---------------------------------------------------------------------------

typedef __attribute__((ext_vector_type(16))) __bf16        v16bf;
typedef __attribute__((ext_vector_type(8)))  float         v8f;
typedef __attribute__((ext_vector_type(4)))  unsigned int  u32x4;

union V16BF { v16bf v; u32x4 q[2]; };
union U4S8  { u32x4 q; unsigned short s[8]; };

__device__ __forceinline__ unsigned short f2bf(float f) {
    union { float f; unsigned u; } v; v.f = f;
    unsigned u = v.u + 0x7FFFu + ((v.u >> 16) & 1u);   // round-to-nearest-even
    return (unsigned short)(u >> 16);
}
__device__ __forceinline__ float bf2f(unsigned short h) {
    union { unsigned u; float f; } v; v.u = ((unsigned)h) << 16; return v.f;
}
__device__ __forceinline__ float silu_f(float y) { return y / (1.0f + __expf(-y)); }
__device__ __forceinline__ float gelu_f(float y) {
    return 0.5f * y * (1.0f + erff(y * 0.70710678118654752f));
}

// CDNA5 async global->LDS copy (16B per lane), tracked by ASYNCcnt.
__device__ __forceinline__ void async_cp_b128(unsigned lds_off, const void* gptr) {
    asm volatile("global_load_async_to_lds_b128 %0, %1, off"
                 :: "v"(lds_off), "v"((unsigned long long)(size_t)gptr)
                 : "memory");
}
__device__ __forceinline__ void wait_async0() {
    asm volatile("s_wait_asynccnt 0x0" ::: "memory");
}

// ---------------------------------------------------------------------------
__global__ void k_f2bf(const float* __restrict__ in, unsigned short* __restrict__ out, int n) {
    int i = blockIdx.x * blockDim.x + threadIdx.x;
    int stride = gridDim.x * blockDim.x;
    for (; i < n; i += stride) out[i] = f2bf(in[i]);
}

__global__ void k_bnfuse(const float* __restrict__ g, const float* __restrict__ b,
                         const float* __restrict__ m, const float* __restrict__ v,
                         float* __restrict__ sc, float* __restrict__ sh, int c) {
    int i = blockIdx.x * blockDim.x + threadIdx.x;
    if (i < c) {
        float s = g[i] * rsqrtf(v[i] + 1e-5f);
        sc[i] = s;
        sh[i] = b[i] - m[i] * s;
    }
}

// ---------------------------------------------------------------------------
// bf16 WMMA GEMM: out[b] = epi( W(MxK) @ X[b](KxN) ), M=256, N=4096, K in {256,512}.
// Block tile 128x128, K-chunk 32, double-buffered LDS, 8 waves -> 4x2 WMMA tiles.
//   epi 0: BN+SiLU -> tokens[b][n][m] bf16 (transposed, 1 b128 store/lane)
//   epi 1: BN+SiLU -> cat[b][256+m][n] bf16
//   epi 2: +bias,GELU,+tokens residual -> cat[b][m][n] bf16 (scale = node_b)
//   epi 3: BN+SiLU -> d_out[b][m][n] f32
// ---------------------------------------------------------------------------
#define TILE_M 128
#define TILE_N 128
#define TILE_K 32

__global__ __launch_bounds__(256) void gemm_bf16(
    const unsigned short* __restrict__ W,
    const unsigned short* __restrict__ X,
    int K, long xBatchStride,
    const float* __restrict__ scale,
    const float* __restrict__ shift,
    const unsigned short* __restrict__ resid,
    void* __restrict__ out, int epi)
{
    const int N = 4096;
    __shared__ __align__(16) unsigned short lA[2][TILE_M * TILE_K];
    __shared__ __align__(16) unsigned short lB[2][TILE_N * TILE_K];

    const int tid  = threadIdx.x;
    const int lane = tid & 31;
    const int wave = tid >> 5;
    const int wr   = wave >> 2;       // 0..1 : 64 rows
    const int wc   = wave & 3;        // 0..3 : 32 cols
    const int b     = blockIdx.z;
    const int mBase = blockIdx.y * TILE_M;
    const int nBase = blockIdx.x * TILE_N;
    const unsigned short* Xb = X + (long)b * xBatchStride;

    // per-thread staging coordinates (A: 512 x b128 straight copy; B: transpose)
    const int aRow0 = tid >> 2,           aC0 = (tid & 3) * 8;
    const int aRow1 = (256 + tid) >> 2,   aC1 = ((256 + tid) & 3) * 8;
    const int bK0   = tid >> 4,           bN0 = (tid & 15) * 8;
    const int bK1   = (256 + tid) >> 4,   bN1 = ((256 + tid) & 15) * 8;

    auto stage = [&](int buf, int k0) {
        unsigned short* lAp = &lA[buf][0];
        unsigned short* lBp = &lB[buf][0];
        // A tile: async DMA directly into LDS (no VGPR round trip)
        async_cp_b128((unsigned)(size_t)(lAp + (size_t)tid * 8),
                      W + (long)(mBase + aRow0) * K + k0 + aC0);
        async_cp_b128((unsigned)(size_t)(lAp + (size_t)(256 + tid) * 8),
                      W + (long)(mBase + aRow1) * K + k0 + aC1);
        // B tile: batch both global b128 loads, then scatter-transpose to [n][k]
        U4S8 t0, t1;
        t0.q = *(const u32x4*)(Xb + (long)(k0 + bK0) * N + nBase + bN0);
        t1.q = *(const u32x4*)(Xb + (long)(k0 + bK1) * N + nBase + bN1);
#pragma unroll
        for (int j = 0; j < 8; ++j) lBp[(bN0 + j) * TILE_K + bK0] = t0.s[j];
#pragma unroll
        for (int j = 0; j < 8; ++j) lBp[(bN1 + j) * TILE_K + bK1] = t1.s[j];
    };

    v8f acc[4][2];
#pragma unroll
    for (int i = 0; i < 4; ++i)
#pragma unroll
        for (int j = 0; j < 2; ++j)
            acc[i][j] = (v8f){0.f,0.f,0.f,0.f,0.f,0.f,0.f,0.f};

    const int lrow = lane & 15;
    const int koff = (lane & 16) ? 8 : 0;   // bf16 fragment K-offset per ISA layout
    const int nk = K / TILE_K;

    stage(0, 0);

    for (int kc = 0; kc < nk; ++kc) {
        const int cur = kc & 1;
        wait_async0();          // my async writes to LDS are visible
        __syncthreads();        // everyone's staging of buf[cur] complete
        if (kc + 1 < nk) stage(cur ^ 1, (kc + 1) * TILE_K);

        V16BF fa[4], fb[2];
#pragma unroll
        for (int mt = 0; mt < 4; ++mt) {
            const unsigned short* p = &lA[cur][(wr * 64 + mt * 16 + lrow) * TILE_K];
            fa[mt].q[0] = *(const u32x4*)(p + koff);
            fa[mt].q[1] = *(const u32x4*)(p + koff + 16);
        }
#pragma unroll
        for (int nt = 0; nt < 2; ++nt) {
            const unsigned short* p = &lB[cur][(wc * 32 + nt * 16 + lrow) * TILE_K];
            fb[nt].q[0] = *(const u32x4*)(p + koff);
            fb[nt].q[1] = *(const u32x4*)(p + koff + 16);
        }
#pragma unroll
        for (int mt = 0; mt < 4; ++mt)
#pragma unroll
            for (int nt = 0; nt < 2; ++nt)
                acc[mt][nt] = __builtin_amdgcn_wmma_f32_16x16x32_bf16(
                    false, fa[mt].v, false, fb[nt].v,
                    (short)0, acc[mt][nt], false, false);
    }

    // ---- epilogue: C/D layout: vgpr r, lane l -> m = r + 8*(l>=16), n = l&15 ----
    const int m_hi = (lane >> 4) * 8;
    const int nl   = lane & 15;
#pragma unroll
    for (int mt = 0; mt < 4; ++mt) {
#pragma unroll
        for (int nt = 0; nt < 2; ++nt) {
            int gm0 = mBase + wr * 64 + mt * 16 + m_hi;   // 8 consecutive m per lane
            int gn  = nBase + wc * 32 + nt * 16 + nl;
            v8f a = acc[mt][nt];
            if (epi == 0) {                       // conv1 -> tokens[b][n][m] bf16
                U4S8 pk;
#pragma unroll
                for (int r = 0; r < 8; ++r)
                    pk.s[r] = f2bf(silu_f(a[r] * scale[gm0 + r] + shift[gm0 + r]));
                unsigned short* dst = (unsigned short*)out + ((long)b * N + gn) * 256 + gm0;
                *(u32x4*)dst = pk.q;
            } else if (epi == 1) {                // conv2 -> cat[b][256+m][n] bf16
                unsigned short* o = (unsigned short*)out;
#pragma unroll
                for (int r = 0; r < 8; ++r) {
                    float y = silu_f(a[r] * scale[gm0 + r] + shift[gm0 + r]);
                    o[((long)b * 512 + 256 + gm0 + r) * N + gn] = f2bf(y);
                }
            } else if (epi == 2) {                // node -> cat[b][m][n] bf16
                U4S8 rz;
                rz.q = *(const u32x4*)(resid + ((long)b * N + gn) * 256 + gm0);
                unsigned short* o = (unsigned short*)out;
#pragma unroll
                for (int r = 0; r < 8; ++r) {
                    float y = gelu_f(a[r] + scale[gm0 + r]);   // scale = node bias
                    y += bf2f(rz.s[r]);                        // + tokens residual
                    o[((long)b * 512 + gm0 + r) * N + gn] = f2bf(y);
                }
            } else {                              // conv3 -> d_out[b][m][n] f32
                float* o = (float*)out;
#pragma unroll
                for (int r = 0; r < 8; ++r)
                    o[((long)b * 256 + gm0 + r) * N + gn] =
                        silu_f(a[r] * scale[gm0 + r] + shift[gm0 + r]);
            }
        }
    }
}

// ---------------------------------------------------------------------------
// ctx avg/max over N, two-stage for parallelism
// ---------------------------------------------------------------------------
__global__ __launch_bounds__(256) void k_ctx_part(const unsigned short* __restrict__ tokens,
                                                  float* __restrict__ psum,
                                                  float* __restrict__ pmax) {
    int ch = blockIdx.x, b = blockIdx.y, c = threadIdx.x;
    float sum = 0.f, mx = -1e30f;
    for (int i = 0; i < 256; ++i) {
        int n = ch * 256 + i;
        float t = bf2f(tokens[((long)b * 4096 + n) * 256 + c]);
        sum += t; mx = fmaxf(mx, t);
    }
    psum[((long)b * 16 + ch) * 256 + c] = sum;
    pmax[((long)b * 16 + ch) * 256 + c] = mx;
}

__global__ __launch_bounds__(256) void k_ctx_final(const float* __restrict__ psum,
                                                   const float* __restrict__ pmax,
                                                   float* __restrict__ ctx) {
    int b = blockIdx.x, c = threadIdx.x;
    float s = 0.f, m = -1e30f;
    for (int ch = 0; ch < 16; ++ch) {
        s += psum[((long)b * 16 + ch) * 256 + c];
        m = fmaxf(m, pmax[((long)b * 16 + ch) * 256 + c]);
    }
    ctx[b * 512 + c]       = s * (1.0f / 4096.0f);
    ctx[b * 512 + 256 + c] = m;
}

// ---------------------------------------------------------------------------
// protos[b] = proto + (ctx @ ctx_w^T + ctx_b);  Q[b] = protos @ pre_w ;
// d[b][e] = pre_b . protos[b][e]   (head-mean collapse of the logit GEMM)
// ---------------------------------------------------------------------------
__global__ __launch_bounds__(256) void k_protoq(
    const float* __restrict__ ctx, const float* __restrict__ ctx_w,
    const float* __restrict__ ctx_b, const float* __restrict__ proto,
    const float* __restrict__ pre_w, const float* __restrict__ pre_b,
    unsigned short* __restrict__ Q, float* __restrict__ dbias) {
    __shared__ float P[8 * 256];
    int b = blockIdx.x, c = threadIdx.x;
    const float* cb = ctx + b * 512;
    for (int e = 0; e < 8; ++e) {
        float o = ctx_b[e * 256 + c];
        const float* wrow = ctx_w + (long)(e * 256 + c) * 512;
        for (int j = 0; j < 512; ++j) o += cb[j] * wrow[j];
        P[e * 256 + c] = proto[e * 256 + c] + o;
    }
    __syncthreads();
    for (int e = 0; e < 8; ++e) {
        float q = 0.f;
        for (int j = 0; j < 256; ++j) q += P[e * 256 + j] * pre_w[j * 256 + c];
        Q[((long)b * 8 + e) * 256 + c] = f2bf(q);
    }
    if (c < 8) {
        float dd = 0.f;
        for (int j = 0; j < 256; ++j) dd += pre_b[j] * P[c * 256 + j];
        dbias[b * 8 + c] = dd;
    }
}

// logits[b][e][n] = (tokens[b][n][:] . Q[b][e][:] + d[b][e]) / 64
__global__ __launch_bounds__(256) void k_logits(const unsigned short* __restrict__ tokens,
                                                const unsigned short* __restrict__ Q,
                                                const float* __restrict__ dbias,
                                                float* __restrict__ logits) {
    __shared__ float Qs[8 * 256];
    __shared__ float ds[8];
    int tid = threadIdx.x;
    int b = blockIdx.x >> 4;
    int n = ((blockIdx.x & 15) << 8) + tid;
    for (int i = tid; i < 2048; i += 256) Qs[i] = bf2f(Q[(long)b * 2048 + i]);
    if (tid < 8) ds[tid] = dbias[b * 8 + tid];
    __syncthreads();
    float acc[8];
#pragma unroll
    for (int e = 0; e < 8; ++e) acc[e] = 0.f;
    const unsigned short* trow = tokens + ((long)b * 4096 + n) * 256;
    for (int c0 = 0; c0 < 256; c0 += 8) {
        U4S8 u; u.q = *(const u32x4*)(trow + c0);
#pragma unroll
        for (int j = 0; j < 8; ++j) {
            float t = bf2f(u.s[j]);
#pragma unroll
            for (int e = 0; e < 8; ++e) acc[e] += t * Qs[e * 256 + c0 + j];
        }
    }
#pragma unroll
    for (int e = 0; e < 8; ++e)
        logits[((long)b * 8 + e) * 4096 + n] = (acc[e] + ds[e]) * (1.0f / 64.0f);
}

// softmax over the 4096 nodes per (b, e); A stored bf16 [b][e][n]
__global__ __launch_bounds__(256) void k_softmax(const float* __restrict__ logits,
                                                 unsigned short* __restrict__ A) {
    __shared__ float red[256];
    int tid = threadIdx.x;
    long base = (long)blockIdx.x * 4096;
    float lv[16];
    float mx = -1e30f;
#pragma unroll
    for (int i = 0; i < 16; ++i) {
        lv[i] = logits[base + tid + i * 256];
        mx = fmaxf(mx, lv[i]);
    }
    red[tid] = mx; __syncthreads();
    for (int s = 128; s > 0; s >>= 1) {
        if (tid < s) red[tid] = fmaxf(red[tid], red[tid + s]);
        __syncthreads();
    }
    mx = red[0]; __syncthreads();
    float sum = 0.f;
#pragma unroll
    for (int i = 0; i < 16; ++i) { lv[i] = __expf(lv[i] - mx); sum += lv[i]; }
    red[tid] = sum; __syncthreads();
    for (int s = 128; s > 0; s >>= 1) {
        if (tid < s) red[tid] += red[tid + s];
        __syncthreads();
    }
    float rs = 1.0f / red[0];
#pragma unroll
    for (int i = 0; i < 16; ++i) A[base + tid + i * 256] = f2bf(lv[i] * rs);
}

// He partials: all 8 hyperedges per pass so tokens are read once
__global__ __launch_bounds__(256) void k_he_part(const unsigned short* __restrict__ A,
                                                 const unsigned short* __restrict__ tokens,
                                                 float* __restrict__ part) {
    int ch = blockIdx.x, b = blockIdx.y, c = threadIdx.x;
    float acc[8];
#pragma unroll
    for (int e = 0; e < 8; ++e) acc[e] = 0.f;
    for (int i = 0; i < 256; ++i) {
        int n = ch * 256 + i;
        float t = bf2f(tokens[((long)b * 4096 + n) * 256 + c]);
#pragma unroll
        for (int e = 0; e < 8; ++e)
            acc[e] += t * bf2f(A[((long)b * 8 + e) * 4096 + n]);
    }
#pragma unroll
    for (int e = 0; e < 8; ++e)
        part[(((long)b * 8 + e) * 16 + ch) * 256 + c] = acc[e];
}

// reduce partials + edge GEMM + gelu
__global__ __launch_bounds__(256) void k_he_final(const float* __restrict__ part,
                                                  const float* __restrict__ edge_w,
                                                  const float* __restrict__ edge_b,
                                                  float* __restrict__ He2) {
    __shared__ float HeL[256];
    int b = blockIdx.x >> 3, e = blockIdx.x & 7, c = threadIdx.x;
    float acc = 0.f;
    for (int ch = 0; ch < 16; ++ch)
        acc += part[(((long)b * 8 + e) * 16 + ch) * 256 + c];
    HeL[c] = acc; __syncthreads();
    float y = edge_b[c];
    const float* wrow = edge_w + (long)c * 256;
    for (int k = 0; k < 256; ++k) y += HeL[k] * wrow[k];
    He2[((long)b * 8 + e) * 256 + c] = gelu_f(y);
}

// Xn[b][c][n] = sum_e He2[b][e][c] * A[b][e][n]   (bf16, K-major for node GEMM)
__global__ __launch_bounds__(256) void k_xn(const float* __restrict__ He2,
                                            const unsigned short* __restrict__ A,
                                            unsigned short* __restrict__ Xn) {
    int b = blockIdx.z, c = blockIdx.y;
    int n = blockIdx.x * 256 + threadIdx.x;
    float s = 0.f;
#pragma unroll
    for (int e = 0; e < 8; ++e)
        s += He2[((long)b * 8 + e) * 256 + c] * bf2f(A[((long)b * 8 + e) * 4096 + n]);
    Xn[((long)b * 256 + c) * 4096 + n] = f2bf(s);
}

// ---------------------------------------------------------------------------
extern "C" void kernel_launch(void* const* d_in, const int* in_sizes, int n_in,
                              void* d_out, int out_size, void* d_ws, size_t ws_size,
                              hipStream_t stream) {
    const int B = 16, C1 = 256, CH = 256, N = 4096, E = 8;

    const float* x      = (const float*)d_in[0];
    const float* cv1_w  = (const float*)d_in[1];
    const float* cv1_g  = (const float*)d_in[2];
    const float* cv1_b  = (const float*)d_in[3];
    const float* cv1_m  = (const float*)d_in[4];
    const float* cv1_v  = (const float*)d_in[5];
    const float* cv2_w  = (const float*)d_in[6];
    const float* cv2_g  = (const float*)d_in[7];
    const float* cv2_b  = (const float*)d_in[8];
    const float* cv2_m  = (const float*)d_in[9];
    const float* cv2_v  = (const float*)d_in[10];
    const float* cv3_w  = (const float*)d_in[11];
    const float* cv3_g  = (const float*)d_in[12];
    const float* cv3_b  = (const float*)d_in[13];
    const float* cv3_m  = (const float*)d_in[14];
    const float* cv3_v  = (const float*)d_in[15];
    const float* proto  = (const float*)d_in[16];
    const float* ctx_w  = (const float*)d_in[17];
    const float* ctx_b  = (const float*)d_in[18];
    const float* pre_w  = (const float*)d_in[19];
    const float* pre_b  = (const float*)d_in[20];
    const float* edge_w = (const float*)d_in[21];
    const float* edge_b = (const float*)d_in[22];
    const float* node_w = (const float*)d_in[23];
    const float* node_b = (const float*)d_in[24];

    char* base = (char*)d_ws;
    size_t off = 0;
    auto carve = [&](size_t bytes) -> void* {
        void* p = base + off;
        off += (bytes + 255) & ~(size_t)255;
        return p;
    };

    unsigned short* xbf    = (unsigned short*)carve((size_t)B * C1 * N * 2); // reused as Xn
    unsigned short* tokens = (unsigned short*)carve((size_t)B * N * CH * 2);
    unsigned short* cat    = (unsigned short*)carve((size_t)B * 512 * N * 2);
    float*          logits = (float*)carve((size_t)B * E * N * 4);
    unsigned short* Abuf   = (unsigned short*)carve((size_t)B * E * N * 2);
    unsigned short* Qbuf   = (unsigned short*)carve((size_t)B * E * CH * 2);
    float*          dbias  = (float*)carve((size_t)B * E * 4);
    float*          ctx    = (float*)carve((size_t)B * 512 * 4);
    float*          He2    = (float*)carve((size_t)B * E * CH * 4);
    float*          psum   = (float*)carve((size_t)B * 16 * 256 * 4);
    float*          pmax   = (float*)carve((size_t)B * 16 * 256 * 4);
    float*          hepart = (float*)carve((size_t)B * E * 16 * 256 * 4);
    unsigned short* w1     = (unsigned short*)carve(65536 * 2);
    unsigned short* w2     = (unsigned short*)carve(65536 * 2);
    unsigned short* w3     = (unsigned short*)carve(131072 * 2);
    unsigned short* wn     = (unsigned short*)carve(65536 * 2);
    float* s1 = (float*)carve(256 * 4); float* h1 = (float*)carve(256 * 4);
    float* s2 = (float*)carve(256 * 4); float* h2 = (float*)carve(256 * 4);
    float* s3 = (float*)carve(256 * 4); float* h3 = (float*)carve(256 * 4);
    (void)ws_size; (void)in_sizes; (void)n_in; (void)out_size;

    // --- precision prep ---
    k_f2bf<<<dim3(16384), 256, 0, stream>>>(x, xbf, B * C1 * N);
    k_f2bf<<<dim3(256),   256, 0, stream>>>(cv1_w, w1, 65536);
    k_f2bf<<<dim3(256),   256, 0, stream>>>(cv2_w, w2, 65536);
    k_f2bf<<<dim3(512),   256, 0, stream>>>(cv3_w, w3, 131072);
    k_f2bf<<<dim3(256),   256, 0, stream>>>(node_w, wn, 65536);
    k_bnfuse<<<dim3(1), 256, 0, stream>>>(cv1_g, cv1_b, cv1_m, cv1_v, s1, h1, 256);
    k_bnfuse<<<dim3(1), 256, 0, stream>>>(cv2_g, cv2_b, cv2_m, cv2_v, s2, h2, 256);
    k_bnfuse<<<dim3(1), 256, 0, stream>>>(cv3_g, cv3_b, cv3_m, cv3_v, s3, h3, 256);

    dim3 ggrid(N / TILE_N, 256 / TILE_M, B);   // (32, 2, 16)

    // conv1: tokens = silu(bn(W1 @ x))^T
    gemm_bf16<<<ggrid, 256, 0, stream>>>(w1, xbf, 256, (long)C1 * N,
                                         s1, h1, nullptr, tokens, 0);
    // conv2: cat[256:512] = silu(bn(W2 @ x))
    gemm_bf16<<<ggrid, 256, 0, stream>>>(w2, xbf, 256, (long)C1 * N,
                                         s2, h2, nullptr, cat, 1);

    // --- HGNN (collapsed-head formulation) ---
    k_ctx_part<<<dim3(16, B), 256, 0, stream>>>(tokens, psum, pmax);
    k_ctx_final<<<dim3(B), 256, 0, stream>>>(psum, pmax, ctx);
    k_protoq<<<dim3(B), 256, 0, stream>>>(ctx, ctx_w, ctx_b, proto, pre_w, pre_b,
                                          Qbuf, dbias);
    k_logits<<<dim3(B * 16), 256, 0, stream>>>(tokens, Qbuf, dbias, logits);
    k_softmax<<<dim3(B * E), 256, 0, stream>>>(logits, Abuf);
    k_he_part<<<dim3(16, B), 256, 0, stream>>>(Abuf, tokens, hepart);
    k_he_final<<<dim3(B * E), 256, 0, stream>>>(hepart, edge_w, edge_b, He2);
    k_xn<<<dim3(16, 256, B), 256, 0, stream>>>(He2, Abuf, xbf /* Xn alias */);

    // node GEMM: cat[0:256] = gelu(node_w @ Xn + node_b) + tokens^T
    gemm_bf16<<<ggrid, 256, 0, stream>>>(wn, xbf, 256, (long)CH * N,
                                         node_b, nullptr, tokens, cat, 2);
    // conv3: out = silu(bn(W3 @ cat))  (f32)
    gemm_bf16<<<ggrid, 256, 0, stream>>>(w3, cat, 512, (long)512 * N,
                                         s3, h3, nullptr, d_out, 3);
}
